// CompensatedSparseLinear_71347996721362
// MI455X (gfx1250) — matrix operand order repaired and checked by reading, not compiled, not run on hardware
//
#include <hip/hip_runtime.h>

// CDNA5 / gfx1250: wave32, WMMA 16x16x32 bf16, f32 accumulate.
typedef __attribute__((ext_vector_type(16))) __bf16 v16bf;
typedef __attribute__((ext_vector_type(4)))  __bf16 v4bf;
typedef __attribute__((ext_vector_type(8)))  float  v8f;

#define TM 128
#define TN 128
#define TK 32
// Each 16x32 subtile stored fragment-ordered: 32 lanes x 16 halves = 512 halves.
#define SUBTILE_HALVES 512

__device__ __forceinline__ void split_store(__bf16* hi, __bf16* lo, int off, float4 v) {
    // bf16 hi/lo split (bf16x3 trick): hi = rne(x), lo = rne(x - hi)
    __bf16 h0 = (__bf16)v.x, h1 = (__bf16)v.y, h2 = (__bf16)v.z, h3 = (__bf16)v.w;
    __bf16 l0 = (__bf16)(v.x - (float)h0);
    __bf16 l1 = (__bf16)(v.y - (float)h1);
    __bf16 l2 = (__bf16)(v.z - (float)h2);
    __bf16 l3 = (__bf16)(v.w - (float)h3);
    v4bf hv = {h0, h1, h2, h3};
    v4bf lv = {l0, l1, l2, l3};
    *(v4bf*)(hi + off) = hv;   // 8B aligned (off is a multiple of 4 halves)
    *(v4bf*)(lo + off) = lv;
}

__global__ __launch_bounds__(256)
void csl_gemm_bf16x3(const float* __restrict__ x,
                     const float* __restrict__ W,
                     const float* __restrict__ bias,
                     const float* __restrict__ dB,
                     float* __restrict__ out,
                     int M, int N, int K)
{
    // 4 x 8KB fragment-ordered LDS tiles (32KB total; 320KB/WGP available)
    __shared__ __bf16 sAhi[8 * SUBTILE_HALVES];
    __shared__ __bf16 sAlo[8 * SUBTILE_HALVES];
    __shared__ __bf16 sBhi[8 * SUBTILE_HALVES];
    __shared__ __bf16 sBlo[8 * SUBTILE_HALVES];

    const int t    = threadIdx.x;       // 0..255  (8 waves of 32)
    const int lane = t & 31;
    const int wave = t >> 5;
    const int wm   = wave & 3;          // 4 wave-rows  -> 2 M-subtiles each (32 rows)
    const int wn   = wave >> 2;         // 2 wave-cols  -> 4 N-subtiles each (64 cols)

    const int m0 = blockIdx.y * TM;
    const int n0 = blockIdx.x * TN;

    const v8f vzero = {0.f, 0.f, 0.f, 0.f, 0.f, 0.f, 0.f, 0.f};
    v8f acc[2][4];
    #pragma unroll
    for (int i = 0; i < 2; ++i)
        #pragma unroll
        for (int j = 0; j < 4; ++j)
            acc[i][j] = vzero;

    for (int kb = 0; kb < K; kb += TK) {
        __syncthreads();   // WAR: previous compute done before restaging

        // ---- Stage A (x tile): 128 rows x 32 k-floats = 1024 float4 slots ----
        #pragma unroll
        for (int s = 0; s < 4; ++s) {
            int slot = t + s * 256;
            int row  = slot >> 3;            // 0..127
            int k0   = (slot & 7) << 2;      // 0..28 step 4
            float4 v = *(const float4*)(x + (size_t)(m0 + row) * K + kb + k0);
            // ISA A-layout (16-bit 16x32): lane = r + 16*((k>>3)&1); half = (k&7) + 8*(k>>4)
            int sub   = row >> 4;
            int r     = row & 15;
            int laneA = r + (((k0 >> 3) & 1) << 4);
            int idx   = (k0 & 7) + ((k0 >> 4) << 3);
            split_store(sAhi, sAlo, sub * SUBTILE_HALVES + laneA * 16 + idx, v);
        }

        // ---- Stage B (W + delta_B tile, fused add): 128 cols x 32 k-floats ----
        #pragma unroll
        for (int s = 0; s < 4; ++s) {
            int slot = t + s * 256;
            int col  = slot >> 3;            // output feature within tile
            int k0   = (slot & 7) << 2;
            size_t g = (size_t)(n0 + col) * K + kb + k0;
            float4 w = *(const float4*)(W  + g);
            float4 d = *(const float4*)(dB + g);
            float4 v = make_float4(w.x + d.x, w.y + d.y, w.z + d.z, w.w + d.w);
            // B-layout (same k order as A): lane = c + 16*((k>>3)&1); half = (k&7) + 8*(k>>4)
            int sub   = col >> 4;
            int c     = col & 15;
            int laneB = c + (((k0 >> 3) & 1) << 4);
            int idx   = (k0 & 7) + ((k0 >> 4) << 3);
            split_store(sBhi, sBlo, sub * SUBTILE_HALVES + laneB * 16 + idx, v);
        }

        __syncthreads();   // RAW: tiles visible before compute

        // ---- Compute: fragments are contiguous 32B per lane ----
        v16bf ahi[2], alo[2], bhi[4], blo[4];
        #pragma unroll
        for (int i = 0; i < 2; ++i) {
            int off = (wm * 2 + i) * SUBTILE_HALVES + lane * 16;
            ahi[i] = *(const v16bf*)(sAhi + off);
            alo[i] = *(const v16bf*)(sAlo + off);
        }
        #pragma unroll
        for (int j = 0; j < 4; ++j) {
            int off = (wn * 4 + j) * SUBTILE_HALVES + lane * 16;
            bhi[j] = *(const v16bf*)(sBhi + off);
            blo[j] = *(const v16bf*)(sBlo + off);
        }

        #pragma unroll
        for (int i = 0; i < 2; ++i) {
            #pragma unroll
            for (int j = 0; j < 4; ++j) {
                // bf16x3: hi*hi + hi*lo + lo*hi (lo*lo below fp32 ulp)
                acc[i][j] = __builtin_amdgcn_wmma_f32_16x16x32_bf16(
                    false, ahi[i], false, bhi[j], (short)0, acc[i][j], false, false);
                acc[i][j] = __builtin_amdgcn_wmma_f32_16x16x32_bf16(
                    false, ahi[i], false, blo[j], (short)0, acc[i][j], false, false);
                acc[i][j] = __builtin_amdgcn_wmma_f32_16x16x32_bf16(
                    false, alo[i], false, bhi[j], (short)0, acc[i][j], false, false);
            }
        }
    }

    // ---- Epilogue: C/D 16x16 f32 layout: lane<16 -> N=lane, M=v ; lane>=16 -> M=v+8
    const int cn = lane & 15;
    const int rh = (lane >> 4) << 3;
    #pragma unroll
    for (int i = 0; i < 2; ++i) {
        int rowb = m0 + (wm * 2 + i) * 16 + rh;
        #pragma unroll
        for (int j = 0; j < 4; ++j) {
            int col  = n0 + (wn * 4 + j) * 16 + cn;
            float bv = bias[col];
            #pragma unroll
            for (int v = 0; v < 8; ++v)
                out[(size_t)(rowb + v) * N + col] = acc[i][j][v] + bv;
        }
    }
}

extern "C" void kernel_launch(void* const* d_in, const int* in_sizes, int n_in,
                              void* d_out, int out_size, void* d_ws, size_t ws_size,
                              hipStream_t stream) {
    const float* x  = (const float*)d_in[0];   // [B,S,D_IN] fp32
    const float* W  = (const float*)d_in[1];   // [D_OUT,D_IN] fp32 (2:4 masked)
    const float* b  = (const float*)d_in[2];   // [D_OUT] fp32
    const float* dB = (const float*)d_in[3];   // [D_OUT,D_IN] fp32
    float* out = (float*)d_out;

    const int N = in_sizes[2];          // D_OUT = 4096
    const int K = in_sizes[1] / N;      // D_IN  = 4096
    const int M = in_sizes[0] / K;      // B*S   = 8192

    dim3 grid(N / TN, M / TM);          // (32, 64)
    csl_gemm_bf16x3<<<grid, 256, 0, stream>>>(x, W, b, dB, out, M, N, K);
}
